// MultiHeadSelfAttention_49134425866762
// MI455X (gfx1250) — matrix operand order repaired and checked by reading, not compiled, not run on hardware
//
#include <hip/hip_runtime.h>

// Problem constants (match reference)
#define BQ   4
#define TT   2048
#define DD   1024
#define HH   16
#define DHH  64
#define NQKV 3072   // 3*D

typedef __attribute__((ext_vector_type(16))) _Float16 v16h;
typedef __attribute__((ext_vector_type(8)))  float    v8f;

#define WMMA_F16(A, B, C) \
  __builtin_amdgcn_wmma_f32_16x16x32_f16(false, (A), false, (B), (short)0, (C), false, false)

#define LDS_OFF(p) ((unsigned int)(unsigned long long)(const void*)(p))

// ---------------------------------------------------------------------------
// Fragment loaders (contiguous k order: element s of half h holds k = 16h + s, wave32)
// A matrix 16x32 f16: lane -> row = lane&15, half h = lane>>4.
//   VGPR r: k = h*16 + 2r + {0,1}
__device__ inline v16h frag_a(const _Float16* __restrict__ base, int stride, int k_base) {
  const int lane = threadIdx.x & 31;
  const int h = lane >> 4;
  const _Float16* p = base + (size_t)(lane & 15) * stride + k_base;
  v16h f;
#pragma unroll
  for (int r = 0; r < 8; ++r) {
    const int k = (h << 4) + (r << 1);
    f[2 * r]     = p[k];
    f[2 * r + 1] = p[k + 1];
  }
  return f;
}

// B matrix 32x16 f16, source stored "per-column rows": elem(col,k) = base[col*stride + k].
//   lane -> col = lane&15; VGPR r: k = h*16 + 2r + {0,1}
__device__ inline v16h frag_bT(const _Float16* __restrict__ base, int stride, int k_base) {
  const int lane = threadIdx.x & 31;
  const int h = lane >> 4;
  const _Float16* p = base + (size_t)(lane & 15) * stride + k_base;
  v16h f;
#pragma unroll
  for (int r = 0; r < 8; ++r) {
    const int k = (h << 4) + (r << 1);
    f[2 * r]     = p[k];
    f[2 * r + 1] = p[k + 1];
  }
  return f;
}

// B matrix 32x16 f16, source stored k-major: elem(col,k) = base[k*stride + col].
__device__ inline v16h frag_b_kmajor(const _Float16* __restrict__ base, int stride,
                                     int col_base, int k_base) {
  const int lane = threadIdx.x & 31;
  const int h = lane >> 4;
  const int col = col_base + (lane & 15);
  v16h f;
#pragma unroll
  for (int r = 0; r < 8; ++r) {
    const int k = k_base + (h << 4) + (r << 1);
    f[2 * r]     = base[(size_t)k * stride + col];
    f[2 * r + 1] = base[(size_t)(k + 1) * stride + col];
  }
  return f;
}

// ---------------------------------------------------------------------------
// Kernel 1: qkv = x @ W_qkv + b_qkv  (M=8192, N=3072, K=1024)
// Block tile 128x128, 8 waves (4 M-strips x 2 N-strips), wave tile 32x64
// (2x4 WMMA tiles), K-step 32.  Software-pipelined: global loads for tile
// k+1 overlap the 8 WMMAs on tile k; double-buffered LDS, 1 barrier/step.
__global__ __launch_bounds__(256) void qkv_gemm(const float* __restrict__ x,
                                                const float* __restrict__ W,
                                                const float* __restrict__ bias,
                                                _Float16* __restrict__ Qo,
                                                _Float16* __restrict__ Ko,
                                                _Float16* __restrict__ Vo) {
  __shared__ _Float16 As[2][128 * 36];  // [row][k]
  __shared__ _Float16 Bs[2][128 * 36];  // [col][k]
  const int tid  = threadIdx.x;
  const int wave = tid >> 5;
  const int wm   = wave & 3;   // 0..3 -> M strip of 32
  const int wn   = wave >> 2;  // 0..1 -> N strip of 64
  const int m0   = blockIdx.y * 128;
  const int n0   = blockIdx.x * 128;

  v8f acc[2][4] = {};
  float4 ra[4], rb[4];

  auto load_tiles = [&](int kk) {
#pragma unroll
    for (int j = 0; j < 4; ++j) {   // A: 128x32 f32
      const int idx = (j * 256 + tid) * 4;
      const int row = idx >> 5, k = idx & 31;
      ra[j] = *reinterpret_cast<const float4*>(x + (size_t)(m0 + row) * DD + kk + k);
    }
#pragma unroll
    for (int j = 0; j < 4; ++j) {   // B: 32x128 f32
      const int idx = (j * 256 + tid) * 4;
      const int k = idx >> 7, n = idx & 127;
      rb[j] = *reinterpret_cast<const float4*>(W + (size_t)(kk + k) * NQKV + n0 + n);
    }
  };
  auto store_tiles = [&](int buf) {
#pragma unroll
    for (int j = 0; j < 4; ++j) {
      const int idx = (j * 256 + tid) * 4;
      const int row = idx >> 5, k = idx & 31;
      _Float16* a = &As[buf][row * 36 + k];
      a[0] = (_Float16)ra[j].x; a[1] = (_Float16)ra[j].y;
      a[2] = (_Float16)ra[j].z; a[3] = (_Float16)ra[j].w;
    }
#pragma unroll
    for (int j = 0; j < 4; ++j) {
      const int idx = (j * 256 + tid) * 4;
      const int k = idx >> 7, n = idx & 127;
      Bs[buf][(n + 0) * 36 + k] = (_Float16)rb[j].x;
      Bs[buf][(n + 1) * 36 + k] = (_Float16)rb[j].y;
      Bs[buf][(n + 2) * 36 + k] = (_Float16)rb[j].z;
      Bs[buf][(n + 3) * 36 + k] = (_Float16)rb[j].w;
    }
  };

  load_tiles(0);
  store_tiles(0);
  __syncthreads();

  int cur = 0;
  for (int kk = 0; kk < DD; kk += 32) {
    const bool have_next = (kk + 32) < DD;
    if (have_next) load_tiles(kk + 32);           // overlaps WMMAs below

    const v16h a0 = frag_a(&As[cur][(wm * 32) * 36], 36, 0);
    const v16h a1 = frag_a(&As[cur][(wm * 32 + 16) * 36], 36, 0);
#pragma unroll
    for (int j = 0; j < 4; ++j) {
      const v16h b = frag_bT(&Bs[cur][(wn * 64 + j * 16) * 36], 36, 0);
      acc[0][j] = WMMA_F16(a0, b, acc[0][j]);
      acc[1][j] = WMMA_F16(a1, b, acc[1][j]);
    }

    if (have_next) store_tiles(cur ^ 1);          // other buffer: no WAR
    __syncthreads();
    cur ^= 1;
  }

  const int lane = tid & 31;
  const int h = lane >> 4, cn = lane & 15;
#pragma unroll
  for (int i = 0; i < 2; ++i) {
#pragma unroll
    for (int j = 0; j < 4; ++j) {
      const int ncol = n0 + wn * 64 + j * 16 + cn;
      const float bv = bias[ncol];
      const int sect = ncol >> 10;      // 0=Q 1=K 2=V
      const int d    = ncol & 1023;
      const int head = d >> 6, dh = d & 63;
      _Float16* dst = (sect == 0) ? Qo : ((sect == 1) ? Ko : Vo);
#pragma unroll
      for (int r = 0; r < 8; ++r) {
        const int m = m0 + wm * 32 + i * 16 + h * 8 + r;
        const int bb = m >> 11, t = m & 2047;   // T = 2048
        dst[(((size_t)(bb * HH + head)) * TT + t) * DHH + dh] =
            (_Float16)(acc[i][j][r] + bv);
      }
    }
  }
}

// ---------------------------------------------------------------------------
// Kernel 2: flash attention. One wave per (16-query tile, b*h). Key tiles of
// 32, double-buffered into LDS with global_load_async_to_lds_b128 (ASYNCcnt).
// smem layout: [buf0 K | buf1 K | buf0 V | buf1 V | P tile]
__global__ __launch_bounds__(32) void attn(const _Float16* __restrict__ Q,
                                           const _Float16* __restrict__ K,
                                           const _Float16* __restrict__ V,
                                           const int* __restrict__ amask,
                                           const int* __restrict__ causal_p,
                                           _Float16* __restrict__ Y) {
  __shared__ _Float16 smem[4 * 2048 + 16 * 34];

  const int qt = blockIdx.x;         // 0..127
  const int bh = blockIdx.y;         // 0..63
  const int b  = bh >> 4;            // H = 16
  const int head = bh & 15;
  const int lane = threadIdx.x & 31;
  const int h = lane >> 4, cn = lane & 15;
  const int causal = causal_p[0];
  const float NEG_INF = -__builtin_inff();

  const _Float16* Qb = Q + (size_t)bh * TT * DHH;
  const _Float16* Kb = K + (size_t)bh * TT * DHH;
  const _Float16* Vb = V + (size_t)bh * TT * DHH;

  // Q tile 16x64 -> two A fragments (d = 0..31, 32..63)
  const v16h q0 = frag_a(Qb + (size_t)qt * 16 * DHH, DHH, 0);
  const v16h q1 = frag_a(Qb + (size_t)qt * 16 * DHH, DHH, 32);

  v8f o[4] = {};
  float mrow[8], lrow[8];
#pragma unroll
  for (int r = 0; r < 8; ++r) { mrow[r] = NEG_INF; lrow[r] = 0.f; }

  const int klimit = causal ? (qt * 16 + 16) : TT;

  // Stage one 32x64 K tile + V tile: 256 b128 chunks each of 8 f16,
  // 8 chunks/lane/matrix -> 16 ASYNCcnt ops per lane per tile.
  auto stage = [&](int buf, int kt) {
    _Float16* kdst = smem + buf * 2048;          // K buffers at 0 / 2048
    _Float16* vdst = smem + 4096 + buf * 2048;   // V buffers at 4096 / 6144
#pragma unroll
    for (int i = 0; i < 8; ++i) {
      const int c = i * 32 + lane;  // chunk id, 8 f16 each (tile is contiguous)
      const _Float16* gk = Kb + (size_t)kt * DHH + c * 8;
      const _Float16* gv = Vb + (size_t)kt * DHH + c * 8;
      const unsigned int lk = LDS_OFF(kdst + c * 8);
      const unsigned int lv = LDS_OFF(vdst + c * 8);
      asm volatile("global_load_async_to_lds_b128 %0, %1, off"
                   :: "v"(lk), "v"(gk) : "memory");
      asm volatile("global_load_async_to_lds_b128 %0, %1, off"
                   :: "v"(lv), "v"(gv) : "memory");
    }
  };

  int pb = 0;
  stage(0, 0);
  for (int kt = 0; kt < klimit; kt += 32) {
    const bool have_next = (kt + 32) < klimit;
    if (have_next) {
      stage(pb ^ 1, kt + 32);
      asm volatile("s_wait_asynccnt 0x10" ::: "memory");  // current tile done
    } else {
      asm volatile("s_wait_asynccnt 0x0" ::: "memory");
    }
    const _Float16* Kcur = smem + pb * 2048;
    const _Float16* Vcur = smem + 4096 + pb * 2048;
    _Float16* Ps = smem + 8192;

    // S = Q K^T for two 16-key subtiles, K fragments from LDS
    v8f s[2] = {};
#pragma unroll
    for (int j = 0; j < 2; ++j) {
      const v16h bk0 = frag_bT(Kcur + j * 16 * DHH, DHH, 0);
      const v16h bk1 = frag_bT(Kcur + j * 16 * DHH, DHH, 32);
      s[j] = WMMA_F16(q0, bk0, s[j]);
      s[j] = WMMA_F16(q1, bk1, s[j]);
    }
    // scale + masks
#pragma unroll
    for (int j = 0; j < 2; ++j) {
      const int key = kt + j * 16 + cn;
      const bool keep = amask[b * TT + key] != 0;
#pragma unroll
      for (int r = 0; r < 8; ++r) {
        const int qrow = qt * 16 + h * 8 + r;
        float v = s[j][r] * 0.125f;   // 1/sqrt(64)
        if (!keep || (causal && key > qrow)) v = NEG_INF;
        s[j][r] = v;
      }
    }
    // online softmax update per row (row = h*8+r; reductions stay in 16-lane half)
#pragma unroll
    for (int r = 0; r < 8; ++r) {
      float v = fmaxf(s[0][r], s[1][r]);
#pragma unroll
      for (int sft = 8; sft >= 1; sft >>= 1)
        v = fmaxf(v, __shfl_xor(v, sft, 32));
      const float mnew = fmaxf(mrow[r], v);
      const bool dead = (mnew == NEG_INF);
      const float p0 = dead ? 0.f : __expf(s[0][r] - mnew);
      const float p1 = dead ? 0.f : __expf(s[1][r] - mnew);
      Ps[(h * 8 + r) * 34 + cn]      = (_Float16)p0;
      Ps[(h * 8 + r) * 34 + 16 + cn] = (_Float16)p1;
      float ps = p0 + p1;
#pragma unroll
      for (int sft = 8; sft >= 1; sft >>= 1)
        ps += __shfl_xor(ps, sft, 32);
      const float alpha = dead ? 1.f : __expf(mrow[r] - mnew);
      lrow[r] = lrow[r] * alpha + ps;
      mrow[r] = mnew;
#pragma unroll
      for (int f = 0; f < 4; ++f) o[f][r] *= alpha;
    }
    __syncthreads();
    const v16h pf = frag_a(Ps, 34, 0);           // P as A fragment 16x32
#pragma unroll
    for (int f = 0; f < 4; ++f) {
      const v16h vf = frag_b_kmajor(Vcur, DHH, f * 16, 0);
      o[f] = WMMA_F16(pf, vf, o[f]);
    }
    __syncthreads();
    pb ^= 1;
  }

  // normalize + write y in (B,T,D) layout, f16
#pragma unroll
  for (int r = 0; r < 8; ++r) {
    const float inv = (lrow[r] > 0.f) ? (1.f / lrow[r]) : 0.f;
    const int t = qt * 16 + h * 8 + r;
#pragma unroll
    for (int f = 0; f < 4; ++f) {
      Y[((size_t)(b * TT + t)) * DD + head * DHH + f * 16 + cn] =
          (_Float16)(o[f][r] * inv);
    }
  }
}

// ---------------------------------------------------------------------------
// Kernel 3: out = y @ W_out + b_out  (M=8192, N=1024, K=1024), fp32 output.
// Same pipelined 128x128 block / 32x64 wave tiling; A is already f16.
__global__ __launch_bounds__(256) void out_gemm(const _Float16* __restrict__ Y,
                                                const float* __restrict__ W,
                                                const float* __restrict__ bias,
                                                float* __restrict__ out) {
  __shared__ _Float16 As[2][128 * 36];
  __shared__ _Float16 Bs[2][128 * 36];
  const int tid  = threadIdx.x;
  const int wave = tid >> 5;
  const int wm   = wave & 3;
  const int wn   = wave >> 2;
  const int m0   = blockIdx.y * 128;
  const int n0   = blockIdx.x * 128;

  v8f acc[2][4] = {};
  float4 ra[2];   // 2 x 8 f16
  float4 rb[4];

  auto load_tiles = [&](int kk) {
#pragma unroll
    for (int j = 0; j < 2; ++j) {   // A: 128x32 f16, 8 f16 per op
      const int idx = (j * 256 + tid) * 8;
      const int row = idx >> 5, k = idx & 31;
      ra[j] = *reinterpret_cast<const float4*>(Y + (size_t)(m0 + row) * DD + kk + k);
    }
#pragma unroll
    for (int j = 0; j < 4; ++j) {   // B: 32x128 f32
      const int idx = (j * 256 + tid) * 4;
      const int k = idx >> 7, n = idx & 127;
      rb[j] = *reinterpret_cast<const float4*>(W + (size_t)(kk + k) * DD + n0 + n);
    }
  };
  auto store_tiles = [&](int buf) {
#pragma unroll
    for (int j = 0; j < 2; ++j) {
      const int idx = (j * 256 + tid) * 8;
      const int row = idx >> 5, k = idx & 31;
      const _Float16* src = reinterpret_cast<const _Float16*>(&ra[j]);
      _Float16* a = &As[buf][row * 36 + k];
#pragma unroll
      for (int e = 0; e < 8; ++e) a[e] = src[e];
    }
#pragma unroll
    for (int j = 0; j < 4; ++j) {
      const int idx = (j * 256 + tid) * 4;
      const int k = idx >> 7, n = idx & 127;
      Bs[buf][(n + 0) * 36 + k] = (_Float16)rb[j].x;
      Bs[buf][(n + 1) * 36 + k] = (_Float16)rb[j].y;
      Bs[buf][(n + 2) * 36 + k] = (_Float16)rb[j].z;
      Bs[buf][(n + 3) * 36 + k] = (_Float16)rb[j].w;
    }
  };

  load_tiles(0);
  store_tiles(0);
  __syncthreads();

  int cur = 0;
  for (int kk = 0; kk < DD; kk += 32) {
    const bool have_next = (kk + 32) < DD;
    if (have_next) load_tiles(kk + 32);

    const v16h a0 = frag_a(&As[cur][(wm * 32) * 36], 36, 0);
    const v16h a1 = frag_a(&As[cur][(wm * 32 + 16) * 36], 36, 0);
#pragma unroll
    for (int j = 0; j < 4; ++j) {
      const v16h b = frag_bT(&Bs[cur][(wn * 64 + j * 16) * 36], 36, 0);
      acc[0][j] = WMMA_F16(a0, b, acc[0][j]);
      acc[1][j] = WMMA_F16(a1, b, acc[1][j]);
    }

    if (have_next) store_tiles(cur ^ 1);
    __syncthreads();
    cur ^= 1;
  }

  const int lane = tid & 31;
  const int h = lane >> 4, cn = lane & 15;
#pragma unroll
  for (int i = 0; i < 2; ++i) {
#pragma unroll
    for (int j = 0; j < 4; ++j) {
      const int ncol = n0 + wn * 64 + j * 16 + cn;
      const float bv = bias[ncol];
#pragma unroll
      for (int r = 0; r < 8; ++r) {
        const int m = m0 + wm * 32 + i * 16 + h * 8 + r;
        out[(size_t)m * DD + ncol] = acc[i][j][r] + bv;
      }
    }
  }
}

// ---------------------------------------------------------------------------
extern "C" void kernel_launch(void* const* d_in, const int* in_sizes, int n_in,
                              void* d_out, int out_size, void* d_ws, size_t ws_size,
                              hipStream_t stream) {
  (void)in_sizes; (void)n_in; (void)out_size; (void)ws_size;
  const float* x      = (const float*)d_in[0];
  const int*   amask  = (const int*)d_in[1];
  const float* Wqkv   = (const float*)d_in[2];
  const float* bqkv   = (const float*)d_in[3];
  const float* Wout   = (const float*)d_in[4];
  const float* bout   = (const float*)d_in[5];
  const int*   causal = (const int*)d_in[6];
  float* out = (float*)d_out;

  const size_t per = (size_t)BQ * HH * TT * DHH;  // 8,388,608 elems
  _Float16* Qb = (_Float16*)d_ws;
  _Float16* Kb = Qb + per;
  _Float16* Vb = Kb + per;
  _Float16* Yb = Vb + per;

  qkv_gemm<<<dim3(NQKV / 128, (BQ * TT) / 128), 256, 0, stream>>>(x, Wqkv, bqkv, Qb, Kb, Vb);
  attn<<<dim3(TT / 16, BQ * HH), 32, 0, stream>>>(Qb, Kb, Vb, amask, causal, Yb);
  out_gemm<<<dim3(DD / 128, (BQ * TT) / 128), 256, 0, stream>>>(Yb, Wout, bout, out);
}
